// MoEPredictor_81123342287343
// MI455X (gfx1250) — matrix operand order repaired and checked by reading, not compiled, not run on hardware
//
#include <hip/hip_runtime.h>

// ---------- problem constants ----------
#define BATCH      8192
#define MODES      6
#define DIM        128
#define NEXP       6
#define FUT2       120          // FUT*2
#define NROWS      (BATCH*MODES)   // 49152
#define TRAJ_OFF   0
#define SCORE_OFF  (BATCH*MODES*FUT2)            // 5,898,240
#define AUX_OFF    (SCORE_OFF + BATCH*MODES)     // 5,947,392
#define PROBS_OFF  (AUX_OFF + 1)                 // 5,947,393

// ---------- workspace layout (bytes) ----------
#define WS_SUMS   0
#define WS_GATE   64                         // float[NROWS] gate weights [B,E]
#define WS_WT1H   (WS_GATE + 196608)         // 196672
#define WS_WT2H   (WS_WT1H + 393216)         // 589888
#define WS_WT3H   (WS_WT2H + 786432)         // 1376320
#define WS_WS1H   (WS_WT3H + 368640)         // 1744960
#define WS_WS2H   (WS_WS1H + 196608)         // 1941568
#define WS_WS3H   (WS_WS2H + 98304)          // 2039872  (+768 -> ~1.95 MB total)

typedef _Float16 h16;
typedef __attribute__((ext_vector_type(16))) _Float16 v16h;
typedef __attribute__((ext_vector_type(8)))  _Float16 v8h;
typedef __attribute__((ext_vector_type(4)))  _Float16 v4h;
typedef __attribute__((ext_vector_type(8)))  float    v8f;

// Branchless exact-GELU: Abramowitz-Stegun 7.1.26 erf (max abs err 1.5e-7).
__device__ __forceinline__ float gelu_fast(float x) {
    float s = x * 0.70710678118654752f;
    float a = fabsf(s);
    float t = __builtin_amdgcn_rcpf(__builtin_fmaf(0.3275911f, a, 1.0f));
    float p = t * __builtin_fmaf(t,
                  __builtin_fmaf(t,
                    __builtin_fmaf(t,
                      __builtin_fmaf(t, 1.061405429f, -1.453152027f),
                      1.421413741f),
                    -0.284496736f),
                  0.254829592f);
    float e = __builtin_amdgcn_exp2f(a * a * -1.4426950408889634f);
    float erfv = __builtin_fmaf(-p, e, 1.0f);
    erfv = copysignf(erfv, s);
    return 0.5f * x * (1.0f + erfv);
}

__device__ __forceinline__ v8f wmma16(v16h a, v16h b, v8f c) {
    return __builtin_amdgcn_wmma_f32_16x16x32_f16(
        false, a, false, b, (short)0, c, false, false);
}

// B fragment (32x16 f16): lane L holds column N = L%16, K in table order (8h+i, 16+8h+i).
__device__ __forceinline__ v16h load_bfrag16(const h16* __restrict__ W, int stride,
                                             int kRow, int col0) {
    int l = kRow & 31;
    const h16* p = W + (size_t)(kRow - l + 8 * (l >> 4)) * stride + col0 + (l & 15);
    v16h r;
#pragma unroll
    for (int i = 0; i < 8; i++) { r[i] = p[(size_t)i * stride]; r[i + 8] = p[(size_t)(16 + i) * stride]; }
    return r;
}

__device__ __forceinline__ v16h load_bfrag16_guard(const h16* __restrict__ W, int stride,
                                                   int kRow, int col0, int nvalid) {
    int l = kRow & 31;
    const h16* p = W + (size_t)(kRow - l + 8 * (l >> 4)) * stride + col0 + (l & 15);
    bool ok = (l & 15) < nvalid;
    v16h r;
#pragma unroll
    for (int i = 0; i < 8; i++) { r[i] = ok ? p[(size_t)i * stride] : (h16)0.0f; r[i + 8] = ok ? p[(size_t)(16 + i) * stride] : (h16)0.0f; }
    return r;
}

// A fragment (16x32 f16) from a global f32 row
__device__ __forceinline__ v16h load_afrag_g(const float* __restrict__ xr, int f, int kb) {
    const float* p0 = xr + f * 32 + kb;
    const float* p1 = xr + f * 32 + 16 + kb;
    float4 a = *(const float4*)(p0);
    float4 b = *(const float4*)(p0 + 4);
    float4 c = *(const float4*)(p1);
    float4 d = *(const float4*)(p1 + 4);
    v16h r;
    r[0]=(h16)a.x; r[1]=(h16)a.y; r[2]=(h16)a.z;  r[3]=(h16)a.w;
    r[4]=(h16)b.x; r[5]=(h16)b.y; r[6]=(h16)b.z;  r[7]=(h16)b.w;
    r[8]=(h16)c.x; r[9]=(h16)c.y; r[10]=(h16)c.z; r[11]=(h16)c.w;
    r[12]=(h16)d.x; r[13]=(h16)d.y; r[14]=(h16)d.z; r[15]=(h16)d.w;
    return r;
}

// A fragment from an LDS activation strip (row stride 256 halves).
__device__ __forceinline__ v16h load_afrag_lds(const h16* rowPtr, int f, int kb) {
    v8h lo = *(const v8h*)(rowPtr + f * 32 + kb);
    v8h hi = *(const v8h*)(rowPtr + f * 32 + 16 + kb);
    v16h r;
#pragma unroll
    for (int i = 0; i < 8; i++) { r[i] = lo[i]; r[i + 8] = hi[i]; }
    return r;
}

__device__ __forceinline__ void store_tile(h16* wlds, v8f v, int m0, int col) {
#pragma unroll
    for (int r = 0; r < 8; r++) wlds[(r + m0) * 256 + col] = (h16)v[r];
}

__device__ __forceinline__ v8f zero8() {
    v8f a;
#pragma unroll
    for (int r = 0; r < 8; r++) a[r] = 0.0f;
    return a;
}

// ===================== weight f32 -> f16 conversion =====================
__global__ void moe_cvt4_kernel(const float* __restrict__ src, h16* __restrict__ dst, int n4) {
    int i = blockIdx.x * blockDim.x + threadIdx.x;
    if (i < n4) {
        float4 v = ((const float4*)src)[i];
        v4h o;
        o[0]=(h16)v.x; o[1]=(h16)v.y; o[2]=(h16)v.z; o[3]=(h16)v.w;
        *(v4h*)(dst + (size_t)i * 4) = o;
    }
}

// ===================== router =====================
__global__ __launch_bounds__(256, 1) void moe_router_kernel(
    const float* __restrict__ X,
    const float* __restrict__ Wr1, const float* __restrict__ br1,
    const float* __restrict__ Wr2, const float* __restrict__ br2,
    const float* __restrict__ Wr3, const float* __restrict__ br3,
    float* __restrict__ outProbs, float* __restrict__ gateW,
    float* __restrict__ sums) {
    int b = blockIdx.x * blockDim.x + threadIdx.x;
    if (b >= BATCH) return;

    float ctx[DIM];
#pragma unroll
    for (int d = 0; d < DIM; d++) ctx[d] = 0.0f;
    const float* xb = X + (size_t)b * MODES * DIM;
    for (int m = 0; m < MODES; m++) {
        const float4* xr = (const float4*)(xb + m * DIM);
#pragma unroll
        for (int q = 0; q < DIM / 4; q++) {
            float4 v = xr[q];
            ctx[4*q] += v.x; ctx[4*q+1] += v.y; ctx[4*q+2] += v.z; ctx[4*q+3] += v.w;
        }
    }
#pragma unroll
    for (int d = 0; d < DIM; d++) ctx[d] *= (1.0f / 6.0f);

    float h2[128];
#pragma unroll
    for (int k = 0; k < 128; k++) h2[k] = 0.0f;
    for (int j = 0; j < 256; j++) {
        float acc = br1[j];
#pragma unroll
        for (int d = 0; d < DIM; d++) acc += ctx[d] * Wr1[d * 256 + j];
        float v = gelu_fast(acc);
#pragma unroll
        for (int k = 0; k < 128; k++) h2[k] += v * Wr2[j * 128 + k];
    }
#pragma unroll
    for (int k = 0; k < 128; k++) h2[k] = gelu_fast(h2[k] + br2[k]);

    float lg[NEXP];
#pragma unroll
    for (int o = 0; o < NEXP; o++) {
        float a = br3[o];
#pragma unroll
        for (int k = 0; k < 128; k++) a += h2[k] * Wr3[k * NEXP + o];
        lg[o] = a;
    }

    float mx = lg[0];
#pragma unroll
    for (int o = 1; o < NEXP; o++) mx = fmaxf(mx, lg[o]);
    float p[NEXP], se = 0.0f;
#pragma unroll
    for (int o = 0; o < NEXP; o++) { p[o] = expf(lg[o] - mx); se += p[o]; }
    float inv = 1.0f / se;
#pragma unroll
    for (int o = 0; o < NEXP; o++) {
        p[o] *= inv;
        outProbs[b * NEXP + o] = p[o];
        atomicAdd(&sums[o], p[o]);
    }

    int i1 = 0;
#pragma unroll
    for (int o = 1; o < NEXP; o++) if (lg[o] > lg[i1]) i1 = o;
    int i2 = (i1 == 0) ? 1 : 0;
#pragma unroll
    for (int o = 0; o < NEXP; o++) if (o != i1 && lg[o] > lg[i2]) i2 = o;
    float e2 = expf(lg[i2] - lg[i1]);
    float sg = 1.0f + e2;
#pragma unroll
    for (int o = 0; o < NEXP; o++) {
        float g = (o == i1) ? (1.0f / sg) : ((o == i2) ? (e2 / sg) : 0.0f);
        gateW[b * NEXP + o] = g;
    }
}

// ===================== aux loss =====================
__global__ void moe_aux_kernel(const float* __restrict__ sums, float* __restrict__ outAux) {
    float ent = 0.0f, l2 = 0.0f;
#pragma unroll
    for (int o = 0; o < NEXP; o++) {
        float avg = sums[o] * (1.0f / (float)BATCH);
        ent += -avg * logf(avg + 1e-8f);
        float d = avg - (1.0f / 6.0f);
        l2 += d * d;
    }
    l2 *= (1.0f / 6.0f);
    outAux[0] = -ent * 0.01f + 0.01f * l2;
}

// ===================== fused experts + combine =====================
// 384 blocks x 128 threads (4 waves); each wave owns TWO 16-row strips so every
// B fragment feeds 2 WMMAs (halves weight-fetch bandwidth per matrix op).
__global__ __launch_bounds__(128, 1) void moe_expert_kernel(
    const float* __restrict__ X,
    const h16* __restrict__ Wt1, const float* __restrict__ bt1,
    const h16* __restrict__ Wt2, const float* __restrict__ bt2,
    const h16* __restrict__ Wt3, const float* __restrict__ bt3,
    const h16* __restrict__ Ws1, const float* __restrict__ bs1,
    const h16* __restrict__ Ws2, const float* __restrict__ bs2,
    const h16* __restrict__ Ws3, const float* __restrict__ bs3,
    const float* __restrict__ gateW,
    float* __restrict__ outTraj, float* __restrict__ outScore) {
    __shared__ __align__(16) h16 smem[8 * 16 * 256];   // 64 KB: 8 strips of [16,256] f16

    const int lane = threadIdx.x & 31;
    const int w    = threadIdx.x >> 5;            // 0..3
    const int n  = lane & 15;
    const int m0 = (lane & 16) ? 8 : 0;
    const int kb = (lane & 16) ? 8 : 0;

    h16* wlds[2];
    int rowBase[2];
#pragma unroll
    for (int s = 0; s < 2; s++) {
        wlds[s]    = smem + (w * 2 + s) * 16 * 256;
        rowBase[s] = blockIdx.x * 128 + w * 32 + s * 16;
    }

    v16h aX[2][4];
#pragma unroll
    for (int s = 0; s < 2; s++) {
        const float* xr = X + (size_t)(rowBase[s] + n) * DIM;
#pragma unroll
        for (int f = 0; f < 4; f++) aX[s][f] = load_afrag_g(xr, f, kb);
    }

    v8f accT[2][8];
#pragma unroll
    for (int s = 0; s < 2; s++)
#pragma unroll
        for (int j = 0; j < 8; j++) accT[s][j] = zero8();
    float scoreAcc[2] = {0.0f, 0.0f};

    for (int e = 0; e < NEXP; e++) {
        const h16* wt1 = Wt1 + (size_t)e * 128 * 256;
        const h16* wt2 = Wt2 + (size_t)e * 256 * 256;
        const h16* wt3 = Wt3 + (size_t)e * 256 * 120;
        const h16* ws1 = Ws1 + (size_t)e * 128 * 128;
        const h16* ws2 = Ws2 + (size_t)e * 128 * 64;
        const h16* ws3 = Ws3 + (size_t)e * 64;

        float gw[2][8];
#pragma unroll
        for (int s = 0; s < 2; s++)
#pragma unroll
            for (int r = 0; r < 8; r++)
                gw[s][r] = gateW[((rowBase[s] + r + m0) / MODES) * NEXP + e];

        // ---- traj layer 1: [32,128] @ [128,256] ----
        for (int j = 0; j < 16; j++) {
            v8f acc0 = zero8(), acc1 = zero8();
#pragma unroll
            for (int k = 0; k < 4; k++) {
                v16h bf = load_bfrag16(wt1, 256, k * 32 + lane, j * 16);
                acc0 = wmma16(aX[0][k], bf, acc0);
                acc1 = wmma16(aX[1][k], bf, acc1);
            }
            float bias = bt1[e * 256 + j * 16 + n];
#pragma unroll
            for (int r = 0; r < 8; r++) { acc0[r] = gelu_fast(acc0[r] + bias);
                                          acc1[r] = gelu_fast(acc1[r] + bias); }
            store_tile(wlds[0], acc0, m0, j * 16 + n);
            store_tile(wlds[1], acc1, m0, j * 16 + n);
        }

        // ---- traj layer 2: [32,256] @ [256,256] ----
        v16h aH[2][8];
#pragma unroll
        for (int s = 0; s < 2; s++)
#pragma unroll
            for (int f = 0; f < 8; f++) aH[s][f] = load_afrag_lds(wlds[s] + n * 256, f, kb);
        for (int j = 0; j < 16; j++) {
            v8f acc0 = zero8(), acc1 = zero8();
#pragma unroll
            for (int k = 0; k < 8; k++) {
                v16h bf = load_bfrag16(wt2, 256, k * 32 + lane, j * 16);
                acc0 = wmma16(aH[0][k], bf, acc0);
                acc1 = wmma16(aH[1][k], bf, acc1);
            }
            float bias = bt2[e * 256 + j * 16 + n];
#pragma unroll
            for (int r = 0; r < 8; r++) { acc0[r] = gelu_fast(acc0[r] + bias);
                                          acc1[r] = gelu_fast(acc1[r] + bias); }
            store_tile(wlds[0], acc0, m0, j * 16 + n);
            store_tile(wlds[1], acc1, m0, j * 16 + n);
        }

        // ---- traj layer 3: [32,256] @ [256,120], gated combine in-register ----
#pragma unroll
        for (int s = 0; s < 2; s++)
#pragma unroll
            for (int f = 0; f < 8; f++) aH[s][f] = load_afrag_lds(wlds[s] + n * 256, f, kb);
#pragma unroll
        for (int j = 0; j < 8; j++) {
            v8f acc0 = zero8(), acc1 = zero8();
#pragma unroll
            for (int k = 0; k < 8; k++) {
                v16h bf = (j == 7) ? load_bfrag16_guard(wt3, 120, k * 32 + lane, 112, 8)
                                   : load_bfrag16(wt3, 120, k * 32 + lane, j * 16);
                acc0 = wmma16(aH[0][k], bf, acc0);
                acc1 = wmma16(aH[1][k], bf, acc1);
            }
            int col = j * 16 + n;
            float bias = (col < FUT2) ? bt3[e * 120 + col] : 0.0f;
#pragma unroll
            for (int r = 0; r < 8; r++) {
                accT[0][j][r] += gw[0][r] * (acc0[r] + bias);
                accT[1][j][r] += gw[1][r] * (acc1[r] + bias);
            }
        }

        // ---- score layer 1: [32,128] @ [128,128] ----
        for (int j = 0; j < 8; j++) {
            v8f acc0 = zero8(), acc1 = zero8();
#pragma unroll
            for (int k = 0; k < 4; k++) {
                v16h bf = load_bfrag16(ws1, 128, k * 32 + lane, j * 16);
                acc0 = wmma16(aX[0][k], bf, acc0);
                acc1 = wmma16(aX[1][k], bf, acc1);
            }
            float bias = bs1[e * 128 + j * 16 + n];
#pragma unroll
            for (int r = 0; r < 8; r++) { acc0[r] = gelu_fast(acc0[r] + bias);
                                          acc1[r] = gelu_fast(acc1[r] + bias); }
            store_tile(wlds[0], acc0, m0, j * 16 + n);
            store_tile(wlds[1], acc1, m0, j * 16 + n);
        }

        // ---- score layer 2: [32,128] @ [128,64] ----
        v16h aS[2][4];
#pragma unroll
        for (int s = 0; s < 2; s++)
#pragma unroll
            for (int f = 0; f < 4; f++) aS[s][f] = load_afrag_lds(wlds[s] + n * 256, f, kb);
        for (int j = 0; j < 4; j++) {
            v8f acc0 = zero8(), acc1 = zero8();
#pragma unroll
            for (int k = 0; k < 4; k++) {
                v16h bf = load_bfrag16(ws2, 64, k * 32 + lane, j * 16);
                acc0 = wmma16(aS[0][k], bf, acc0);
                acc1 = wmma16(aS[1][k], bf, acc1);
            }
            float bias = bs2[e * 64 + j * 16 + n];
#pragma unroll
            for (int r = 0; r < 8; r++) { acc0[r] = gelu_fast(acc0[r] + bias);
                                          acc1[r] = gelu_fast(acc1[r] + bias); }
            store_tile(wlds[0], acc0, m0, j * 16 + n);
            store_tile(wlds[1], acc1, m0, j * 16 + n);
        }

        // ---- score layer 3: [32,64] @ [64,1] dot ----
        if (lane < 16) {
#pragma unroll
            for (int s = 0; s < 2; s++) {
                const h16* s2p = wlds[s] + lane * 256;
                float dot = 0.0f;
#pragma unroll
                for (int k2 = 0; k2 < 64; k2++) dot += (float)s2p[k2] * (float)ws3[k2];
                int row = rowBase[s] + lane;
                float g = gateW[(row / MODES) * NEXP + e];
                scoreAcc[s] += g * (dot + bs3[e]);
            }
        }
    }

#pragma unroll
    for (int s = 0; s < 2; s++) {
#pragma unroll
        for (int j = 0; j < 8; j++) {
            int col = j * 16 + n;
            if (col < FUT2) {
#pragma unroll
                for (int r = 0; r < 8; r++) {
                    int row = rowBase[s] + r + m0;
                    outTraj[(size_t)row * FUT2 + col] = accT[s][j][r];
                }
            }
        }
        if (lane < 16) outScore[rowBase[s] + lane] = scoreAcc[s];
    }
}

// ===================== launch =====================
extern "C" void kernel_launch(void* const* d_in, const int* in_sizes, int n_in,
                              void* d_out, int out_size, void* d_ws, size_t ws_size,
                              hipStream_t stream) {
    const float* X   = (const float*)d_in[0];
    const float* Wr1 = (const float*)d_in[1];  const float* br1 = (const float*)d_in[2];
    const float* Wr2 = (const float*)d_in[3];  const float* br2 = (const float*)d_in[4];
    const float* Wr3 = (const float*)d_in[5];  const float* br3 = (const float*)d_in[6];
    const float* Wt1 = (const float*)d_in[7];  const float* bt1 = (const float*)d_in[8];
    const float* Wt2 = (const float*)d_in[9];  const float* bt2 = (const float*)d_in[10];
    const float* Wt3 = (const float*)d_in[11]; const float* bt3 = (const float*)d_in[12];
    const float* Ws1 = (const float*)d_in[13]; const float* bs1 = (const float*)d_in[14];
    const float* Ws2 = (const float*)d_in[15]; const float* bs2 = (const float*)d_in[16];
    const float* Ws3 = (const float*)d_in[17]; const float* bs3 = (const float*)d_in[18];

    float* out      = (float*)d_out;
    float* outTraj  = out + TRAJ_OFF;
    float* outScore = out + SCORE_OFF;
    float* outAux   = out + AUX_OFF;
    float* outProbs = out + PROBS_OFF;

    char* ws = (char*)d_ws;
    float* sums  = (float*)(ws + WS_SUMS);
    float* gateW = (float*)(ws + WS_GATE);
    h16* wt1h = (h16*)(ws + WS_WT1H);
    h16* wt2h = (h16*)(ws + WS_WT2H);
    h16* wt3h = (h16*)(ws + WS_WT3H);
    h16* ws1h = (h16*)(ws + WS_WS1H);
    h16* ws2h = (h16*)(ws + WS_WS2H);
    h16* ws3h = (h16*)(ws + WS_WS3H);

    hipMemsetAsync(d_ws, 0, 64, stream);

    {
        struct { const float* s; h16* d; int n; } cv[6] = {
            { Wt1, wt1h, NEXP*128*256 }, { Wt2, wt2h, NEXP*256*256 },
            { Wt3, wt3h, NEXP*256*120 }, { Ws1, ws1h, NEXP*128*128 },
            { Ws2, ws2h, NEXP*128*64 },  { Ws3, ws3h, NEXP*64 },
        };
        for (int i = 0; i < 6; i++) {
            int n4 = cv[i].n / 4;
            moe_cvt4_kernel<<<(n4 + 255) / 256, 256, 0, stream>>>(cv[i].s, cv[i].d, n4);
        }
    }

    moe_router_kernel<<<BATCH / 256, 256, 0, stream>>>(
        X, Wr1, br1, Wr2, br2, Wr3, br3, outProbs, gateW, sums);

    moe_aux_kernel<<<1, 1, 0, stream>>>(sums, outAux);

    moe_expert_kernel<<<NROWS / 128, 128, 0, stream>>>(
        X, wt1h, bt1, wt2h, bt2, wt3h, bt3, ws1h, bs1, ws2h, bs2, ws3h, bs3,
        gateW, outTraj, outScore);
}
